// DisparityNet_55791625175658
// MI455X (gfx1250) — matrix operand order repaired and checked by reading, not compiled, not run on hardware
//
#include <hip/hip_runtime.h>
#include <hip/hip_bf16.h>
#include <math.h>

// ---------------------------------------------------------------------------
// DisparityNet on MI455X (gfx1250, wave32, WMMA)
//
// cost[w,d] = |fl(w)|^2 + |fr(w-d)|^2 - 2 * <fl(w), fr(w-d)>
// The cross term is computed as 16x16x32 f16 WMMA GEMMs (M=16 pixels,
// K=32 channels, N=16 fr positions); 7 N-tiles cover the 112-wide window
// needed for D=96 disparities per 16-pixel tile.
// ---------------------------------------------------------------------------

typedef _Float16 h8  __attribute__((ext_vector_type(8)));
typedef _Float16 v16h __attribute__((ext_vector_type(16)));
typedef float    v8f  __attribute__((ext_vector_type(8)));

#define BN   2
#define HH   384
#define WW   1280
#define CC   3
#define FF   32
#define DD   96
#define HH2  192
#define WW2  640

// ---------------------------------------------------------------------------
// Kernel 1: 3x3 stride-2 SAME conv (NHWC, HWIO) + ReLU, output f16 half-res.
// block = 256 threads = 8 pixels x 32 output channels; weights in LDS.
// JAX SAME for (in=384/1280, k=3, s=2): pad_lo=0, pad_hi=1 -> ih = 2*oh + kh.
// ---------------------------------------------------------------------------
__global__ __launch_bounds__(256) void conv_relu_f16(
    const float* __restrict__ xl, const float* __restrict__ xr,
    const float* __restrict__ Wt, const float* __restrict__ bias,
    _Float16* __restrict__ feat_half) {
  __shared__ float sW[3 * 3 * CC * FF];   // 864
  __shared__ float sB[FF];
  int t = threadIdx.x;
  for (int i = t; i < 3 * 3 * CC * FF; i += 256) sW[i] = Wt[i];
  if (t < FF) sB[t] = bias[t];
  __syncthreads();

  int f  = t & 31;
  int pg = t >> 5;
  int pixel = blockIdx.x * 8 + pg;              // 0 .. 4*192*640-1
  int ow = pixel % WW2;
  int oh = (pixel / WW2) % HH2;
  int s  = pixel / (WW2 * HH2);                 // 0..3 (0,1=left 2,3=right)
  const float* x = (s < BN) ? xl : xr;
  int b = (s < BN) ? s : s - BN;

  float acc = sB[f];
  for (int kh = 0; kh < 3; ++kh) {
    int ih = oh * 2 + kh;
    if (ih >= HH) continue;
    for (int kw = 0; kw < 3; ++kw) {
      int iw = ow * 2 + kw;
      if (iw >= WW) continue;
      const float* xp = x + (((long)b * HH + ih) * WW + iw) * CC;
      const float* wp = sW + ((kh * 3 + kw) * CC) * FF + f;
      acc += xp[0] * wp[0] + xp[1] * wp[FF] + xp[2] * wp[2 * FF];
    }
  }
  acc = fmaxf(acc, 0.0f);
  feat_half[(((long)s * HH2 + oh) * WW2 + ow) * FF + f] = (_Float16)acc;
}

// ---------------------------------------------------------------------------
// Kernel 2: half-pixel bilinear 2x upsample (edge clamp, matches
// jax.image.resize "bilinear") -> f16 full-res features + f32 squared norm
// per pixel (norm computed from the f16-rounded values so it matches what
// the WMMA dot products see). One thread per output pixel, 32 channels.
// ---------------------------------------------------------------------------
__global__ __launch_bounds__(256) void upsample_norm(
    const _Float16* __restrict__ feat_half,
    _Float16* __restrict__ feat_full,
    float* __restrict__ norms) {
  long n = (long)blockIdx.x * 256 + threadIdx.x; // 0 .. 4*384*1280-1
  int ow = (int)(n % WW);
  int oh = (int)((n / WW) % HH);
  int s  = (int)(n / ((long)WW * HH));

  // out coord i maps to input i/2 - 0.25: even -> {k-1:0.25, k:0.75},
  // odd -> {k:0.75, k+1:0.25}
  int hk = oh >> 1, wk = ow >> 1;
  int h0, w0; float wy0, wx0;
  if (oh & 1) { h0 = hk;     wy0 = 0.75f; } else { h0 = hk - 1; wy0 = 0.25f; }
  if (ow & 1) { w0 = wk;     wx0 = 0.75f; } else { w0 = wk - 1; wx0 = 0.25f; }
  int h1 = h0 + 1; if (h1 > HH2 - 1) h1 = HH2 - 1;
  if (h0 < 0) h0 = 0;
  int w1 = w0 + 1; if (w1 > WW2 - 1) w1 = WW2 - 1;
  if (w0 < 0) w0 = 0;
  float wy1 = 1.0f - wy0, wx1 = 1.0f - wx0;

  const _Float16* base = feat_half + (long)s * HH2 * WW2 * FF;
  const h8* p00 = (const h8*)(base + ((long)h0 * WW2 + w0) * FF);
  const h8* p01 = (const h8*)(base + ((long)h0 * WW2 + w1) * FF);
  const h8* p10 = (const h8*)(base + ((long)h1 * WW2 + w0) * FF);
  const h8* p11 = (const h8*)(base + ((long)h1 * WW2 + w1) * FF);
  h8* outp = (h8*)(feat_full + n * FF);

  float nrm = 0.0f;
  for (int j = 0; j < 4; ++j) {
    h8 a = p00[j], bq = p01[j], c = p10[j], d = p11[j];
    h8 o;
    for (int e = 0; e < 8; ++e) {
      float v = wy0 * (wx0 * (float)a[e] + wx1 * (float)bq[e]) +
                wy1 * (wx0 * (float)c[e] + wx1 * (float)d[e]);
      _Float16 hv = (_Float16)v;
      o[e] = hv;
      float fv = (float)hv;
      nrm += fv * fv;
    }
    outp[j] = o;
  }
  norms[n] = nrm;
}

// lane-pair reduction helper: ds_swizzle SWAPX1 (xor=1, and=0x1f) -> 0x041f
__device__ __forceinline__ float swz_xor1(float x) {
  return __int_as_float(__builtin_amdgcn_ds_swizzle(__float_as_int(x), 0x041f));
}

// ---------------------------------------------------------------------------
// Kernel 3: fused cost volume + soft-argmin.
// One block = one (b,h) row x 64-pixel tile. 4 waves; wave i owns the 16
// pixels [wbase+16i, wbase+16i+15]. fr window for the block:
// wbase-96 .. wbase+63 (160 positions, circular in W).
// Each wave: A = fl tile (16x32 f16), then 7 WMMAs over 16-wide fr N-tiles
// -> G[m][0..111] = <fl(w0+m), fr(w0-96+j)> staged in LDS.
// cost[m][d] = nl[m] + nr[j] - 2*G[m][j],  j = m - d + 96.
// Soft-argmin: s = -177*cost/max_d(cost); softmax-stable via min cost.
// d-dimension split across thread pairs, combined with ds_swizzle.
// ---------------------------------------------------------------------------
__global__ __launch_bounds__(128) void cost_softargmin(
    const _Float16* __restrict__ feat_full,
    const float* __restrict__ norms,
    float* __restrict__ out) {
  __shared__ _Float16 s_fl[64][FF];     //  4 KB
  __shared__ _Float16 s_fr[160][FF];    // 10 KB
  __shared__ float    s_nl[64];
  __shared__ float    s_nr[160];
  __shared__ float    s_G[4][16][112];  // 28 KB

  int t = threadIdx.x;
  int wtile = blockIdx.x % (WW / 64);             // 0..19
  int h     = (blockIdx.x / (WW / 64)) % HH;
  int b     = blockIdx.x / ((WW / 64) * HH);      // 0..1
  int wbase = wtile * 64;

  const _Float16* fl = feat_full + (((long)b * HH + h) * WW) * FF;
  const _Float16* fr = feat_full + (((long)(b + BN) * HH + h) * WW) * FF;
  const float* nlp = norms + ((long)b * HH + h) * WW;
  const float* nrp = norms + ((long)(b + BN) * HH + h) * WW;

  // ---- stage fl (64 rows x 4 x h8), fr (160 rows x 4 x h8), norms ----
  for (int j = t; j < 64 * 4; j += 128) {
    int row = j >> 2, ch = j & 3;
    *(h8*)&s_fl[row][ch * 8] =
        *(const h8*)(fl + (long)(wbase + row) * FF + ch * 8);
  }
  for (int j = t; j < 160 * 4; j += 128) {
    int row = j >> 2, ch = j & 3;
    int w = wbase - 96 + row; if (w < 0) w += WW;   // circular roll
    *(h8*)&s_fr[row][ch * 8] = *(const h8*)(fr + (long)w * FF + ch * 8);
  }
  if (t < 64) s_nl[t] = nlp[wbase + t];
  for (int j = t; j < 160; j += 128) {
    int w = wbase - 96 + j; if (w < 0) w += WW;
    s_nr[j] = nrp[w];
  }
  __syncthreads();

  // ---- WMMA phase (EXEC all-ones here) ----
  int wave = t >> 5;
  int lane = t & 31;
  int ml   = lane & 15;
  int hi   = lane >> 4;

  // A fragment (16x32 f16, ISA layout): lane<16 holds K {0..7,16..23},
  // lane>=16 holds K {8..15,24..31}, row M = lane&15.
  v16h afrag;
  {
    int kb0 = hi * 8;
    h8 lo  = *(const h8*)&s_fl[wave * 16 + ml][kb0];
    h8 hi8 = *(const h8*)&s_fl[wave * 16 + ml][kb0 + 16];
    afrag = (v16h)__builtin_shufflevector(lo, hi8,
        0, 1, 2, 3, 4, 5, 6, 7, 8, 9, 10, 11, 12, 13, 14, 15);
  }

#pragma unroll
  for (int tt = 0; tt < 7; ++tt) {
    // B fragment (32x16 f16): column N = lane&15, K {0..7,16..23} (lanes 0-15) /
    // K {8..15,24..31} (lanes 16-31), same K order as A.
    int rowB = wave * 16 + tt * 16 + ml;
    int kbB  = hi * 8;
    h8 blo = *(const h8*)&s_fr[rowB][kbB];
    h8 bhi = *(const h8*)&s_fr[rowB][kbB + 16];
    v16h bfrag = (v16h)__builtin_shufflevector(blo, bhi,
        0, 1, 2, 3, 4, 5, 6, 7, 8, 9, 10, 11, 12, 13, 14, 15);

    v8f c = {};
    c = __builtin_amdgcn_wmma_f32_16x16x32_f16(
        /*neg_a=*/false, afrag, /*neg_b=*/false, bfrag,
        /*c_mod=*/(short)0, c, /*reuse_a=*/false, /*reuse_b=*/false);

    // C/D layout: lane<16 -> N=lane, M=r; lane>=16 -> N=lane-16, M=r+8
#pragma unroll
    for (int r = 0; r < 8; ++r)
      s_G[wave][hi * 8 + r][tt * 16 + ml] = c[r];
  }
  __syncthreads();

  // ---- soft-argmin: thread pair (2p, 2p+1) handles pixel p, 48 d each ----
  int p     = t >> 1;        // 0..63
  int halfd = t & 1;
  int wv = p >> 4, m = p & 15;
  float nlv = s_nl[p];

  float cmin = 3.4e38f, cmax = -3.4e38f;
  for (int k = 0; k < 48; ++k) {
    int d  = halfd * 48 + k;
    int jj = m - d + 96;                       // 1..111
    float cost = nlv + s_nr[wv * 16 + jj] - 2.0f * s_G[wv][m][jj];
    cmin = fminf(cmin, cost);
    cmax = fmaxf(cmax, cost);
  }
  cmin = fminf(cmin, swz_xor1(cmin));
  cmax = fmaxf(cmax, swz_xor1(cmax));

  float scale = -177.0f / cmax;                // s = scale * cost
  float smax  = scale * cmin;                  // max_d s (min cost)
  float esum = 0.0f, wsum = 0.0f;
  for (int k = 0; k < 48; ++k) {
    int d  = halfd * 48 + k;
    int jj = m - d + 96;
    float cost = nlv + s_nr[wv * 16 + jj] - 2.0f * s_G[wv][m][jj];
    float e = __expf(scale * cost - smax);
    esum += e;
    wsum += (float)d * e;
  }
  esum += swz_xor1(esum);
  wsum += swz_xor1(wsum);

  if (halfd == 0)
    out[((long)b * HH + h) * WW + wbase + p] = wsum / esum;
}

// ---------------------------------------------------------------------------
// launcher
// ---------------------------------------------------------------------------
extern "C" void kernel_launch(void* const* d_in, const int* in_sizes, int n_in,
                              void* d_out, int out_size, void* d_ws,
                              size_t ws_size, hipStream_t stream) {
  (void)in_sizes; (void)n_in; (void)out_size; (void)ws_size;
  const float* xl   = (const float*)d_in[0];
  const float* xr   = (const float*)d_in[1];
  const float* Wt   = (const float*)d_in[2];
  const float* bias = (const float*)d_in[3];
  // d_in[4] = num_disparity == 96 (fixed shape, compiled in)

  // workspace layout (256B-aligned regions):
  //   feat_half: 4*192*640*32 f16   = 31,457,280 B
  //   feat_full: 4*384*1280*32 f16  = 125,829,120 B
  //   norms:     4*384*1280  f32    = 7,864,320 B
  char* ws = (char*)d_ws;
  _Float16* feat_half = (_Float16*)ws;
  _Float16* feat_full = (_Float16*)(ws + 31457280);
  float*    nrm       = (float*)(ws + 31457280 + 125829120);

  conv_relu_f16<<<(4 * HH2 * WW2) / 8, 256, 0, stream>>>(xl, xr, Wt, bias,
                                                         feat_half);
  upsample_norm<<<(4 * HH * WW) / 256, 256, 0, stream>>>(feat_half, feat_full,
                                                         nrm);
  cost_softargmin<<<BN * HH * (WW / 64), 128, 0, stream>>>(feat_full, nrm,
                                                           (float*)d_out);
}
